// DecoderAttention_72249939853877
// MI455X (gfx1250) — matrix operand order repaired and checked by reading, not compiled, not run on hardware
//
#include <hip/hip_runtime.h>
#include <hip/hip_bf16.h>

// ---------- types ----------
typedef __bf16 bf16;
typedef __attribute__((ext_vector_type(16))) __bf16 v16bf;
typedef __attribute__((ext_vector_type(8)))  __bf16 v8bf;
typedef __attribute__((ext_vector_type(2)))  __bf16 v2bf;
typedef __attribute__((ext_vector_type(8)))  float  v8f;
typedef __attribute__((ext_vector_type(4)))  float  v4f;
typedef __attribute__((ext_vector_type(4)))  unsigned int v4u;
typedef __attribute__((ext_vector_type(4)))  unsigned int sg4;   // D# group0
typedef __attribute__((ext_vector_type(8)))  unsigned int sg8;   // D# group1

// Problem constants
#define B_   32
#define LQ_  256
#define LK_  2048
#define D_   512
#define H_   8

// ---------- fp32 -> bf16 (prefer v_cvt_pk_bf16_f32) ----------
__device__ __forceinline__ unsigned int pack2(float a, float b) {
#if __has_builtin(__builtin_amdgcn_cvt_pk_bf16_f32)
  v2bf p = __builtin_amdgcn_cvt_pk_bf16_f32(a, b);
  return __builtin_bit_cast(unsigned int, p);
#else
  union { float f; unsigned int u; } x, y; x.f = a; y.f = b;
  unsigned int ra = x.u + 0x7FFFu + ((x.u >> 16) & 1u);
  unsigned int rb = y.u + 0x7FFFu + ((y.u >> 16) & 1u);
  return (ra >> 16) | (rb & 0xFFFF0000u);
#endif
}
__device__ __forceinline__ bf16 f2bf(float f) {
#if __has_builtin(__builtin_amdgcn_cvt_pk_bf16_f32)
  v2bf p = __builtin_amdgcn_cvt_pk_bf16_f32(f, f);
  return p[0];
#else
  union { float f; unsigned int u; } a; a.f = f;
  unsigned int r = a.u + 0x7FFFu + ((a.u >> 16) & 1u);
  return __builtin_bit_cast(__bf16, (unsigned short)(r >> 16));
#endif
}

__device__ __forceinline__ v8f wmma_bf16(v16bf a, v16bf b, v8f c) {
  return __builtin_amdgcn_wmma_f32_16x16x32_bf16(false, a, false, b, (short)0, c,
                                                 false, false);
}

// ---------- 16-lane xor-butterfly via v_permlane16_b32 (VALU, no LDS) ----------
#if __has_builtin(__builtin_amdgcn_permlane16)
__device__ __forceinline__ float px(float v, unsigned int lo, unsigned int hi) {
  int i = __builtin_bit_cast(int, v);
  int r = __builtin_amdgcn_permlane16(i, i, (int)lo, (int)hi, true, false);
  return __builtin_bit_cast(float, r);
}
__device__ __forceinline__ float redmax16(float v) {
  v = fmaxf(v, px(v, 0x67452301u, 0xEFCDAB89u));   // xor 1
  v = fmaxf(v, px(v, 0x54761032u, 0xDCFE98BAu));   // xor 2
  v = fmaxf(v, px(v, 0x32107654u, 0xBA98FEDCu));   // xor 4
  v = fmaxf(v, px(v, 0xFEDCBA98u, 0x76543210u));   // xor 8
  return v;
}
__device__ __forceinline__ float redsum16(float v) {
  v += px(v, 0x67452301u, 0xEFCDAB89u);
  v += px(v, 0x54761032u, 0xDCFE98BAu);
  v += px(v, 0x32107654u, 0xBA98FEDCu);
  v += px(v, 0xFEDCBA98u, 0x76543210u);
  return v;
}
#else
__device__ __forceinline__ float redmax16(float v) {
  #pragma unroll
  for (int m = 1; m < 16; m <<= 1) v = fmaxf(v, __shfl_xor(v, m, 32));
  return v;
}
__device__ __forceinline__ float redsum16(float v) {
  #pragma unroll
  for (int m = 1; m < 16; m <<= 1) v += __shfl_xor(v, m, 32);
  return v;
}
#endif

// ---------------------------------------------------------------------
// Tensor Data Mover: 2D tile load (global -> LDS) with per-row pad.
// ---------------------------------------------------------------------
__device__ __forceinline__ void tdm_load_2d(unsigned long long gaddr, unsigned int lds_addr,
                                            unsigned int tensor_d0, unsigned int tensor_d1,
                                            unsigned int tile_d0, unsigned int tile_d1,
                                            unsigned int stride_d0,
                                            unsigned int pad_int_code, unsigned int pad_amt_code) {
  sg4 g0;
  g0[0] = 1u;                                              // count = 1 (valid)
  g0[1] = lds_addr;                                        // LDS byte address
  g0[2] = (unsigned int)gaddr;                             // global_addr[31:0]
  g0[3] = ((unsigned int)(gaddr >> 32) & 0x01FFFFFFu)      // global_addr[56:32]
          | (2u << 30);                                    // type = 2
  sg8 g1;
  g1[0] = (1u << 16)                                       // data_size = 2 bytes
        | (1u << 20)                                       // pad_enable
        | (pad_int_code << 22) | (pad_amt_code << 25);
  g1[1] = (tensor_d0 & 0xFFFFu) << 16;
  g1[2] = (tensor_d0 >> 16) | ((tensor_d1 & 0xFFFFu) << 16);
  g1[3] = (tensor_d1 >> 16) | (tile_d0 << 16);
  g1[4] = tile_d1;
  g1[5] = stride_d0;
  g1[6] = 0u;
  g1[7] = 0u;
  asm volatile("tensor_load_to_lds %0, %1" :: "s"(g0), "s"(g1) : "memory");
}

// =====================================================================
// Generic projection GEMM: Y(M,512) = X(M,512) @ W(512,512)^T + bias
// =====================================================================
template <int MODE>
__global__ __launch_bounds__(256)
void gemm_bias_kernel(const float* __restrict__ X, const float* __restrict__ W,
                      const float* __restrict__ bias, bf16* __restrict__ Y) {
  __shared__ __align__(16) bf16 Xs[128 * 40];

  const int tid  = threadIdx.x;
  const int lane = tid & 31;
  const int wave = tid >> 5;
  const int wm = wave >> 1, wn = wave & 1;
  const int m0 = blockIdx.x * 128;
  const int n0 = blockIdx.y * 64;
  const int l16 = lane & 15, half = lane >> 4;

  const int lr = tid >> 1;
  const int lc = (tid & 1) * 16;

  v8f acc[2][2] = {};

  for (int k0 = 0; k0 < 512; k0 += 32) {
    {
      const float* src = X + (size_t)(m0 + lr) * 512 + k0 + lc;
      v4f f0 = *(const v4f*)(src);
      v4f f1 = *(const v4f*)(src + 4);
      v4f f2v = *(const v4f*)(src + 8);
      v4f f3 = *(const v4f*)(src + 12);
      float fv[16];
      #pragma unroll
      for (int i = 0; i < 4; i++) {
        fv[i] = f0[i]; fv[4 + i] = f1[i]; fv[8 + i] = f2v[i]; fv[12 + i] = f3[i];
      }
      unsigned int pk[8];
      #pragma unroll
      for (int i = 0; i < 8; i++) pk[i] = pack2(fv[2 * i], fv[2 * i + 1]);
      v4u s0 = {pk[0], pk[1], pk[2], pk[3]};
      v4u s1 = {pk[4], pk[5], pk[6], pk[7]};
      *(v4u*)(&Xs[lr * 40 + lc])     = s0;
      *(v4u*)(&Xs[lr * 40 + lc + 8]) = s1;
    }
    __syncthreads();

    v16bf a[2];
    #pragma unroll
    for (int s = 0; s < 2; s++) {
      const int row = wm * 32 + s * 16 + l16;
      const int kb  = half * 16;
      v8bf lo = *(const v8bf*)(&Xs[row * 40 + kb]);
      v8bf hi = *(const v8bf*)(&Xs[row * 40 + kb + 8]);
      #pragma unroll
      for (int i = 0; i < 8; i++) { a[s][i] = lo[i]; a[s][i + 8] = hi[i]; }
    }
    v16bf bfr[2];
    #pragma unroll
    for (int s = 0; s < 2; s++) {
      const int col = n0 + wn * 32 + s * 16 + l16;
      const float* wp = W + (size_t)col * 512 + k0 + half * 16;
      v4f g0 = *(const v4f*)wp;       v4f g1 = *(const v4f*)(wp + 4);
      v4f g2 = *(const v4f*)(wp + 8); v4f g3 = *(const v4f*)(wp + 12);
      unsigned int w0 = pack2(g0[0], g0[1]), w1 = pack2(g0[2], g0[3]);
      unsigned int w2 = pack2(g1[0], g1[1]), w3 = pack2(g1[2], g1[3]);
      unsigned int w4 = pack2(g2[0], g2[1]), w5 = pack2(g2[2], g2[3]);
      unsigned int w6 = pack2(g3[0], g3[1]), w7 = pack2(g3[2], g3[3]);
      v4u p0 = {w0, w1, w2, w3};
      v4u p1 = {w4, w5, w6, w7};
      v8bf lo = __builtin_bit_cast(v8bf, p0);
      v8bf hi = __builtin_bit_cast(v8bf, p1);
      #pragma unroll
      for (int i = 0; i < 8; i++) { bfr[s][i] = lo[i]; bfr[s][i + 8] = hi[i]; }
    }
    #pragma unroll
    for (int i = 0; i < 2; i++)
      #pragma unroll
      for (int j = 0; j < 2; j++)
        acc[i][j] = wmma_bf16(a[i], bfr[j], acc[i][j]);
    __syncthreads();
  }

  #pragma unroll
  for (int i = 0; i < 2; i++) {
    #pragma unroll
    for (int j = 0; j < 2; j++) {
      const int col = n0 + wn * 32 + j * 16 + l16;
      const float bv = bias[col];
      #pragma unroll
      for (int r = 0; r < 8; r++) {
        const int row = m0 + wm * 32 + i * 16 + r + half * 8;
        const float v = acc[i][j][r] + bv;
        if (MODE == 0) {
          Y[(size_t)row * 512 + col] = f2bf(v);
        } else {
          const int bb = row >> 11, jj = row & 2047;
          const int hh = col >> 6,  dd = col & 63;
          Y[((size_t)((bb << 3) | hh) * 64 + dd) * 2048 + jj] = f2bf(v);
        }
      }
    }
  }
}

// =====================================================================
// Flash attention, TDM-fed, 64-key chunks:
//  - one workgroup = 128 queries of one (b,h): 8 waves x 16 queries
//  - wave 0 drives the TDM: double-buffered K (64x64) and V (64x64)
//    tiles into LDS (32-DW rows + 4-DW TDM pad -> 144B stride,
//    conflict-free b128 fragment reads), s_wait_tensorcnt + barriers.
//  - per-chunk: 8 WMMA (QK^T) + 8 WMMA (PV); one softmax round / 64 keys
//  - grid = B*H*(LQ/128) = 512 blocks.
// =====================================================================
__global__ __launch_bounds__(256)
void flash_attn_kernel(const bf16* __restrict__ qp, const bf16* __restrict__ kp,
                       const bf16* __restrict__ vpT,
                       const unsigned char* __restrict__ mask,
                       float* __restrict__ ctx) {
  __shared__ __align__(16) bf16 Kbuf[2][64 * 72];   // 18432 B
  __shared__ __align__(16) bf16 Vbuf[2][64 * 72];   // 18432 B
  __shared__ __align__(16) bf16 Ps[8][16 * 72];     // 18432 B

  const int tid  = threadIdx.x;
  const int lane = tid & 31;
  const int wave = tid >> 5;
  const int blk  = blockIdx.x;
  const int qh = blk & 1;
  const int h  = (blk >> 1) & 7;
  const int b  = blk >> 4;
  const int l16 = lane & 15, half = lane >> 4;
  const int q0 = qh * 128 + wave * 16;

  const unsigned long long kgbase =
      (unsigned long long)(size_t)kp + ((size_t)(b * 2048) * 512 + h * 64) * 2;
  const unsigned long long vgbase =
      (unsigned long long)(size_t)vpT + ((size_t)((b * 8 + h) * 64) * 2048) * 2;
  const unsigned int klds0 = (unsigned int)(size_t)(&Kbuf[0][0]);
  const unsigned int klds1 = (unsigned int)(size_t)(&Kbuf[1][0]);
  const unsigned int vlds0 = (unsigned int)(size_t)(&Vbuf[0][0]);
  const unsigned int vlds1 = (unsigned int)(size_t)(&Vbuf[1][0]);

  // Q fragments: 16 queries x 64 dims -> two 16x32 A fragments
  v16bf aq[2];
  {
    const bf16* qrow = qp + (size_t)(b * 256 + q0 + l16) * 512 + h * 64;
    #pragma unroll
    for (int kk = 0; kk < 2; kk++) {
      const int kb = kk * 32 + half * 16;
      v8bf lo = *(const v8bf*)(qrow + kb);
      v8bf hi = *(const v8bf*)(qrow + kb + 8);
      #pragma unroll
      for (int i = 0; i < 8; i++) { aq[kk][i] = lo[i]; aq[kk][i + 8] = hi[i]; }
    }
  }

  float mrow[8], lrow[8];
  v8f o[4] = {};
  #pragma unroll
  for (int r = 0; r < 8; r++) { mrow[r] = -1e30f; lrow[r] = 0.f; }

  const unsigned char* mp = mask + (size_t)b * 2048;
  bf16* myPs = &Ps[wave][0];

  // prologue: prefetch chunk 0 (K rows 64 dims, V rows 64 keys; 32DW + 4DW pad)
  if (wave == 0) {
    tdm_load_2d(kgbase, klds0, 512u, 2048u, 64u, 64u, 512u, 4u, 3u);
    tdm_load_2d(vgbase, vlds0, 2048u, 64u, 64u, 64u, 2048u, 4u, 3u);
  }

  for (int kc = 0; kc < 32; kc++) {
    const int key0 = kc * 64;
    const int buf  = kc & 1;

    if (wave == 0) {
      if (kc < 31) {
        const unsigned long long kg = kgbase + (unsigned long long)(kc + 1) * (64ull * 512ull * 2ull);
        const unsigned long long vg = vgbase + (unsigned long long)(kc + 1) * 128ull;
        const unsigned int kl = (kc & 1) ? klds0 : klds1;
        const unsigned int vl = (kc & 1) ? vlds0 : vlds1;
        tdm_load_2d(kg, kl, 512u, 2048u, 64u, 64u, 512u, 4u, 3u);
        tdm_load_2d(vg, vl, 2048u, 64u, 64u, 64u, 2048u, 4u, 3u);
        asm volatile("s_wait_tensorcnt 0x2" ::: "memory");
      } else {
        asm volatile("s_wait_tensorcnt 0x0" ::: "memory");
      }
    }
    __syncthreads();

    // ---- S = Q K^T (16 x 64 keys) ----
    v8f s[4] = {};
    #pragma unroll
    for (int sub = 0; sub < 4; sub++) {
      const int key = sub * 16 + l16;
      const bf16* krow = &Kbuf[buf][key * 72];
      #pragma unroll
      for (int kk = 0; kk < 2; kk++) {
        const int kb = kk * 32 + half * 16;
        v8bf lo = *(const v8bf*)(krow + kb);
        v8bf hi = *(const v8bf*)(krow + kb + 8);
        v16bf bk;
        #pragma unroll
        for (int i = 0; i < 8; i++) { bk[i] = lo[i]; bk[i + 8] = hi[i]; }
        s[sub] = wmma_bf16(aq[kk], bk, s[sub]);
      }
    }
    // ---- scale + key-padding mask ----
    #pragma unroll
    for (int sub = 0; sub < 4; sub++) {
      const bool msk = mp[key0 + sub * 16 + l16] != 0;
      #pragma unroll
      for (int r = 0; r < 8; r++) {
        const float v = s[sub][r] * 0.125f;
        s[sub][r] = msk ? -1e30f : v;
      }
    }
    // ---- online softmax: one round per 64 keys ----
    float alpha[8];
    #pragma unroll
    for (int r = 0; r < 8; r++) {
      float loc = fmaxf(fmaxf(s[0][r], s[1][r]), fmaxf(s[2][r], s[3][r]));
      const float mx = redmax16(loc);
      const float nm = fmaxf(mrow[r], mx);
      alpha[r] = __expf(mrow[r] - nm);
      mrow[r] = nm;
      s[0][r] = __expf(s[0][r] - nm);
      s[1][r] = __expf(s[1][r] - nm);
      s[2][r] = __expf(s[2][r] - nm);
      s[3][r] = __expf(s[3][r] - nm);
      const float rs = redsum16((s[0][r] + s[1][r]) + (s[2][r] + s[3][r]));
      lrow[r] = lrow[r] * alpha[r] + rs;
    }
    #pragma unroll
    for (int j = 0; j < 4; j++)
      #pragma unroll
      for (int r = 0; r < 8; r++) o[j][r] *= alpha[r];

    // ---- C-layout -> A-layout via per-wave LDS round trip (16x64) ----
    #pragma unroll
    for (int sub = 0; sub < 4; sub++) {
      const int col = sub * 16 + l16;
      #pragma unroll
      for (int r = 0; r < 8; r++) myPs[(r + half * 8) * 72 + col] = f2bf(s[sub][r]);
    }
    asm volatile("s_wait_dscnt 0x0" ::: "memory");
    v16bf ap[2];
    #pragma unroll
    for (int kk = 0; kk < 2; kk++) {
      const int kb = kk * 32 + half * 16;
      v8bf lo = *(const v8bf*)(&myPs[l16 * 72 + kb]);
      v8bf hi = *(const v8bf*)(&myPs[l16 * 72 + kb + 8]);
      #pragma unroll
      for (int i = 0; i < 8; i++) { ap[kk][i] = lo[i]; ap[kk][i + 8] = hi[i]; }
    }
    // ---- O += P @ V ----
    #pragma unroll
    for (int j = 0; j < 4; j++) {
      const int dim = j * 16 + l16;
      #pragma unroll
      for (int kk = 0; kk < 2; kk++) {
        const bf16* vrow = &Vbuf[buf][dim * 72 + kk * 32 + half * 16];
        v8bf lo = *(const v8bf*)vrow;
        v8bf hi = *(const v8bf*)(vrow + 8);
        v16bf bv;
        #pragma unroll
        for (int i = 0; i < 8; i++) { bv[i] = lo[i]; bv[i + 8] = hi[i]; }
        o[j] = wmma_bf16(ap[kk], bv, o[j]);
      }
    }

    asm volatile("s_wait_dscnt 0x0" ::: "memory");
    __syncthreads();
  }

  // ---- normalize and write ctx ----
  #pragma unroll
  for (int j = 0; j < 4; j++) {
    #pragma unroll
    for (int r = 0; r < 8; r++) {
      const int row = q0 + r + half * 8;
      ctx[(size_t)(b * 256 + row) * 512 + h * 64 + j * 16 + l16] = o[j][r] / lrow[r];
    }
  }
}

// =====================================================================
// Final score: out(b,256,2048) = 10*tanh( nc @ k2^T / sqrt(512) ), masked
// =====================================================================
__global__ __launch_bounds__(256)
void score_kernel(const bf16* __restrict__ nc, const bf16* __restrict__ k2,
                  const unsigned char* __restrict__ mask,
                  float* __restrict__ out) {
  const int blk = blockIdx.x;
  const int ng = blk & 7;
  const int mt = (blk >> 3) & 7;
  const int b  = blk >> 6;
  const int wave = threadIdx.x >> 5, lane = threadIdx.x & 31;
  const int l16 = lane & 15, half = lane >> 4;
  const int m0 = mt * 32;
  const int n0 = ng * 256 + wave * 32;

  v8f acc[2][2] = {};
  for (int k0 = 0; k0 < 512; k0 += 32) {
    v16bf a[2], bb[2];
    #pragma unroll
    for (int s = 0; s < 2; s++) {
      const bf16* ar = nc + (size_t)(b * 256 + m0 + s * 16 + l16) * 512 + k0 + half * 16;
      v8bf lo = *(const v8bf*)ar;
      v8bf hi = *(const v8bf*)(ar + 8);
      #pragma unroll
      for (int i = 0; i < 8; i++) { a[s][i] = lo[i]; a[s][i + 8] = hi[i]; }
    }
    #pragma unroll
    for (int s = 0; s < 2; s++) {
      const bf16* br = k2 + (size_t)(b * 2048 + n0 + s * 16 + l16) * 512 + k0 + half * 16;
      v8bf lo = *(const v8bf*)br;
      v8bf hi = *(const v8bf*)(br + 8);
      #pragma unroll
      for (int i = 0; i < 8; i++) { bb[s][i] = lo[i]; bb[s][i + 8] = hi[i]; }
    }
    #pragma unroll
    for (int i = 0; i < 2; i++)
      #pragma unroll
      for (int j = 0; j < 2; j++) acc[i][j] = wmma_bf16(a[i], bb[j], acc[i][j]);
  }

  const float scale = 0.04419417382415922f;   // 1/sqrt(512)
  #pragma unroll
  for (int i = 0; i < 2; i++) {
    #pragma unroll
    for (int j = 0; j < 2; j++) {
      const int col = n0 + j * 16 + l16;
      const bool msk = mask[(size_t)b * 2048 + col] != 0;
      #pragma unroll
      for (int r = 0; r < 8; r++) {
        const int row = m0 + i * 16 + r + half * 8;
        float v = 10.0f * tanhf(acc[i][j][r] * scale);
        if (msk) v = -3.4e38f;
        out[((size_t)b * 256 + row) * 2048 + col] = v;
      }
    }
  }
}

// =====================================================================
extern "C" void kernel_launch(void* const* d_in, const int* in_sizes, int n_in,
                              void* d_out, int out_size, void* d_ws, size_t ws_size,
                              hipStream_t stream) {
  (void)in_sizes; (void)n_in; (void)out_size; (void)ws_size;
  const float* q   = (const float*)d_in[0];
  const float* k   = (const float*)d_in[1];
  const unsigned char* mask = (const unsigned char*)d_in[2];
  const float* Wq  = (const float*)d_in[3];
  const float* bq  = (const float*)d_in[4];
  const float* Wk  = (const float*)d_in[5];
  const float* bk  = (const float*)d_in[6];
  const float* Wv  = (const float*)d_in[7];
  const float* bv  = (const float*)d_in[8];
  const float* Wo  = (const float*)d_in[9];
  const float* bo  = (const float*)d_in[10];
  const float* Wks = (const float*)d_in[11];
  const float* bks = (const float*)d_in[12];

  char* ws = (char*)d_ws;
  bf16*  qp  = (bf16*)ws;  ws += (size_t)B_ * LQ_ * D_ * 2;
  bf16*  kp  = (bf16*)ws;  ws += (size_t)B_ * LK_ * D_ * 2;
  bf16*  vpT = (bf16*)ws;  ws += (size_t)B_ * LK_ * D_ * 2;
  bf16*  k2  = (bf16*)ws;  ws += (size_t)B_ * LK_ * D_ * 2;
  float* ctx = (float*)ws; ws += (size_t)B_ * LQ_ * D_ * 4;
  bf16*  nc  = (bf16*)ws;

  const dim3 blk(256);
  gemm_bias_kernel<0><<<dim3(64, 8),  blk, 0, stream>>>(q,   Wq,  bq,  qp);
  gemm_bias_kernel<0><<<dim3(512, 8), blk, 0, stream>>>(k,   Wk,  bk,  kp);
  gemm_bias_kernel<1><<<dim3(512, 8), blk, 0, stream>>>(k,   Wv,  bv,  vpT);
  gemm_bias_kernel<0><<<dim3(512, 8), blk, 0, stream>>>(k,   Wks, bks, k2);
  flash_attn_kernel<<<dim3(512), blk, 0, stream>>>(qp, kp, vpT, mask, ctx);
  gemm_bias_kernel<0><<<dim3(64, 8), blk, 0, stream>>>(ctx, Wo, bo, nc);
  score_kernel<<<dim3(2048), blk, 0, stream>>>(nc, k2, mask, (float*)d_out);
}
